// MultiheadSelfAttention_75874892251733
// MI455X (gfx1250) — matrix operand order repaired and checked by reading, not compiled, not run on hardware
//
#include <hip/hip_runtime.h>

typedef __attribute__((ext_vector_type(16))) _Float16 v16h;
typedef __attribute__((ext_vector_type(8)))  float    v8f;
typedef __attribute__((ext_vector_type(4)))  unsigned int u32x4;
typedef __attribute__((ext_vector_type(8)))  int          i32x8;
typedef __attribute__((ext_vector_type(4)))  int          i32x4;

#if defined(__has_builtin)
#if __has_builtin(__builtin_amdgcn_tensor_load_to_lds) && __has_builtin(__builtin_amdgcn_s_wait_tensorcnt)
#define HAVE_TDM 1
#endif
#endif

// ---------------------------------------------------------------------------
// WMMA fragment loaders (wave32; A/B in contiguous k order, same for both)
// A (16x32, MxK): lane = half_id*16 + m ; holds row m, K = half_id*16 + {0..15}
// B (32x16, KxN): lane = half_id*16 + j ; holds col j, K = half_id*16 + {0..15}
// C/D (16x16):    lanes 0-15: vgpr r -> M=r,  N=lane ; lanes 16-31: M=8+r
// ---------------------------------------------------------------------------
__device__ __forceinline__ v16h load_a_frag(const _Float16* __restrict__ base,
                                            int lda, int row0, int k0) {
  int lane = threadIdx.x & 31;
  int hi = lane >> 4;
  int m  = lane & 15;
  const _Float16* r = base + (long)(row0 + m) * lda + k0 + hi * 16;
  v16h a;
#pragma unroll
  for (int i = 0; i < 8; ++i) { a[i] = r[i]; a[i + 8] = r[8 + i]; }
  return a;
}

__device__ __forceinline__ v16h load_b_frag(const _Float16* __restrict__ wt,
                                            int ldb, int n0, int k0) {
  int lane = threadIdx.x & 31;
  int hi = lane >> 4;
  int j  = lane & 15;
  const _Float16* r = wt + (long)(n0 + j) * ldb + k0 + hi * 16;
  v16h b;
#pragma unroll
  for (int i = 0; i < 16; ++i) b[i] = r[i];
  return b;
}

// ---------------------------------------------------------------------------
// GEMM: out = A(MxK,f16) x BT(NxK,f16)^T + bias; optional relu, f16/f32 out.
// Each wave owns a 16x64 strip (4 accumulators, A-fragment reused 4x).
// Block = 8 waves -> 128 rows x 64 cols. Grid: (M/128, N/64, batch).
// ---------------------------------------------------------------------------
__global__ void gemm_wmma_kernel(const _Float16* __restrict__ A, long sA, int lda,
                                 const _Float16* __restrict__ BT, int ldb,
                                 const float* __restrict__ bias,
                                 _Float16* __restrict__ outh, long sOh, int ldoh,
                                 float* __restrict__ outf, long sOf, int ldof,
                                 int K, int relu) {
  int lane = threadIdx.x & 31;
  int w = threadIdx.x >> 5;
  int tr = blockIdx.x * 128 + w * 16;
  int tc = blockIdx.y * 64;
  const _Float16* Ab = A + sA * blockIdx.z;
  v8f acc[4] = {};
  for (int k0 = 0; k0 < K; k0 += 32) {
    v16h a = load_a_frag(Ab, lda, tr, k0);
    // speculative prefetch of the next K-slice (global_prefetch_b8)
    __builtin_prefetch(Ab + (long)(tr + (lane & 15)) * lda + k0 + 32, 0, 1);
    __builtin_prefetch(BT + (long)(tc + lane) * ldb + k0 + 32, 0, 1);
#pragma unroll
    for (int c = 0; c < 4; ++c) {
      v16h b = load_b_frag(BT, ldb, tc + c * 16, k0);
      acc[c] = __builtin_amdgcn_wmma_f32_16x16x32_f16(false, a, false, b,
                                                      (short)0, acc[c],
                                                      false, false);
    }
  }
  int hi = lane >> 4, nn = lane & 15;
#pragma unroll
  for (int c = 0; c < 4; ++c) {
    int col = tc + c * 16 + nn;
    float bv = bias ? bias[col] : 0.0f;
#pragma unroll
    for (int r = 0; r < 8; ++r) {
      int row = tr + hi * 8 + r;
      float v = acc[c][r] + bv;
      if (relu) v = fmaxf(v, 0.0f);
      if (outh) outh[sOh * blockIdx.z + (long)row * ldoh + col] = (_Float16)v;
      if (outf) outf[sOf * blockIdx.z + (long)row * ldof + col] = v;
    }
  }
}

// ---------------------------------------------------------------------------
// Flash attention, one (batch, head) per blockIdx.{z,y}; 8 waves x 16 q-rows.
// hd = 32 == WMMA K. K-tile fetched by the Tensor Data Mover (wave 0 issues,
// s_wait_tensorcnt, then workgroup barrier); V manually transposed into LDS
// so the PV B-fragment is contiguous. Online softmax on 16-lane halves.
// Epilogue applies the reference's reshape permutation:
//   out[b,h,row,c] -> att[b, h*512 + row/8, (row%8)*32 + c]
// ---------------------------------------------------------------------------
__global__ void flash_attn_kernel(const _Float16* __restrict__ qh,
                                  const _Float16* __restrict__ kh,
                                  const _Float16* __restrict__ vh,
                                  _Float16* __restrict__ att) {
  const int N = 4096, D = 256, HD = 32;
  __shared__ _Float16 lds_k[32 * 32];      // [key][hd]
  __shared__ _Float16 lds_vt[32 * 32];     // [hd][key]  (transposed)
  __shared__ _Float16 lds_p[8][16 * 32];   // per-wave P tile, [row][key]

  int lane = threadIdx.x & 31;
  int w = threadIdx.x >> 5;
  int head = blockIdx.y;
  int batch = blockIdx.z;
  int q0 = blockIdx.x * 128 + w * 16;
  long base = (long)batch * N * D + head * HD;

  // Q tile lives in registers for the whole key sweep (A operand, K = hd = 32)
  v16h Aq = load_a_frag(qh + base, D, q0, 0);

  int hi = lane >> 4, j = lane & 15;
  float mrow[8], lrow[8];
  v8f O0 = {}, O1 = {};
#pragma unroll
  for (int r = 0; r < 8; ++r) { mrow[r] = -1e30f; lrow[r] = 0.0f; }

  int t = threadIdx.x;
  int krow = t >> 3;
  int kc4 = (t & 7) * 4;

#ifdef HAVE_TDM
  // Tensor DMA descriptor (D#): 2D tile, 32 rows x 32 halfs, stride 256 halfs
  u32x4 g0 = {};
  i32x8 g1 = {};
  i32x4 z4 = {};
#if __clang_major__ >= 23
  i32x8 z8 = {};
#endif
  if (w == 0) {
    g0[0] = 1u;                                    // count=1 (valid, user mode)
    g0[1] = (unsigned)(unsigned long long)lds_k;   // lds_addr
    g1[0] = (int)(1u << 16);                       // data_size=1 (2 bytes)
    g1[1] = (int)(256u << 16);                     // tensor_dim0[15:0] = 256
    g1[2] = (int)(4096u << 16);                    // tensor_dim1[15:0] = 4096
    g1[3] = (int)(32u << 16);                      // tile_dim0 = 32
    g1[4] = 32;                                    // tile_dim1 = 32
    g1[5] = 256;                                   // tensor_dim0_stride = 256
    g1[6] = 0;
    g1[7] = 0;
  }
#endif

  for (int kb = 0; kb < N / 32; ++kb) {
    int key0 = kb * 32;
#ifdef HAVE_TDM
    if (w == 0) {
      unsigned long long ga =
          (unsigned long long)(kh + base + (long)key0 * D);
      g0[2] = (unsigned)ga;                                        // addr lo
      g0[3] = (unsigned)((ga >> 32) & 0x01FFFFFFull) | (2u << 30); // hi|type=2
#if __clang_major__ >= 23
      __builtin_amdgcn_tensor_load_to_lds(g0, g1, z4, z4, z8, 0);
#else
      __builtin_amdgcn_tensor_load_to_lds(g0, g1, z4, z4, 0);
#endif
      __builtin_amdgcn_s_wait_tensorcnt(0);
    }
#else
    {
      const _Float16* ksrc = kh + base + (long)(key0 + krow) * D + kc4;
#pragma unroll
      for (int i = 0; i < 4; ++i) lds_k[krow * 32 + kc4 + i] = ksrc[i];
    }
#endif
    const _Float16* vsrc = vh + base + (long)(key0 + krow) * D + kc4;
#pragma unroll
    for (int i = 0; i < 4; ++i) lds_vt[(kc4 + i) * 32 + krow] = vsrc[i];
    __syncthreads();

    // S = Q x K^T : columns are keys; B-fragment rows contiguous in lds_k
    v16h bS0, bS1;
#pragma unroll
    for (int i = 0; i < 16; ++i) {
      bS0[i] = lds_k[j * 32 + hi * 16 + i];
      bS1[i] = lds_k[(j + 16) * 32 + hi * 16 + i];
    }
    v8f z = {};
    v8f S0 = __builtin_amdgcn_wmma_f32_16x16x32_f16(false, Aq, false, bS0,
                                                    (short)0, z, false, false);
    v8f S1 = __builtin_amdgcn_wmma_f32_16x16x32_f16(false, Aq, false, bS1,
                                                    (short)0, z, false, false);

    // online softmax: rows live across a 16-lane half; xor masks 8..1 stay
    // within the half on wave32
#pragma unroll
    for (int r = 0; r < 8; ++r) {
      float mx = fmaxf(S0[r], S1[r]);
#pragma unroll
      for (int off = 8; off >= 1; off >>= 1)
        mx = fmaxf(mx, __shfl_xor(mx, off, 32));
      float mnew = fmaxf(mrow[r], mx);
      float alpha = __expf(mrow[r] - mnew);
      float p0 = __expf(S0[r] - mnew);
      float p1 = __expf(S1[r] - mnew);
      float rs = p0 + p1;
#pragma unroll
      for (int off = 8; off >= 1; off >>= 1) rs += __shfl_xor(rs, off, 32);
      lrow[r] = lrow[r] * alpha + rs;
      mrow[r] = mnew;
      O0[r] *= alpha;
      O1[r] *= alpha;
      // C-layout -> row-major P in per-wave LDS (re-read as A operand)
      lds_p[w][(hi * 8 + r) * 32 + j] = (_Float16)p0;
      lds_p[w][(hi * 8 + r) * 32 + j + 16] = (_Float16)p1;
    }

    // O += P x V : B-fragment contiguous because V is transposed in LDS
    v16h Ap = load_a_frag(&lds_p[w][0], 32, 0, 0);
    v16h bV0, bV1;
#pragma unroll
    for (int i = 0; i < 16; ++i) {
      bV0[i] = lds_vt[j * 32 + hi * 16 + i];
      bV1[i] = lds_vt[(j + 16) * 32 + hi * 16 + i];
    }
    O0 = __builtin_amdgcn_wmma_f32_16x16x32_f16(false, Ap, false, bV0,
                                                (short)0, O0, false, false);
    O1 = __builtin_amdgcn_wmma_f32_16x16x32_f16(false, Ap, false, bV1,
                                                (short)0, O1, false, false);
    __syncthreads();
  }

  // normalize + store with the reference reshape permutation
#pragma unroll
  for (int r = 0; r < 8; ++r) {
    float inv = 1.0f / lrow[r];
    int row = q0 + hi * 8 + r;
    long im = (long)head * 512 + (row >> 3);
    int jb = (row & 7) * 32;
    long ob = (long)batch * N * D + im * D + jb;
    att[ob + j] = (_Float16)(O0[r] * inv);
    att[ob + 16 + j] = (_Float16)(O1[r] * inv);
  }
}

// ---------------------------------------------------------------------------
// LayerNorm over C=256; one wave per row (8 floats/lane). Optionally writes
// f16 into a strided slot (concat buffer) and/or f32 with the transposed-x
// residual (final output).
// ---------------------------------------------------------------------------
__global__ void layernorm_kernel(const float* __restrict__ inp,
                                 const float* __restrict__ g,
                                 const float* __restrict__ b,
                                 _Float16* __restrict__ outh, int ldoh, int coloff,
                                 float* __restrict__ outf,
                                 const float* __restrict__ xsrc) {
  const int C = 256, N = 4096;
  int lane = threadIdx.x & 31;
  int w = threadIdx.x >> 5;
  long row = (long)blockIdx.x * 8 + w;
  const float* rp = inp + row * C + lane * 8;
  float v[8];
  float s = 0.0f;
#pragma unroll
  for (int i = 0; i < 8; ++i) { v[i] = rp[i]; s += v[i]; }
#pragma unroll
  for (int off = 16; off >= 1; off >>= 1) s += __shfl_xor(s, off, 32);
  float mu = s * (1.0f / 256.0f);
  float q = 0.0f;
#pragma unroll
  for (int i = 0; i < 8; ++i) { float d = v[i] - mu; q += d * d; }
#pragma unroll
  for (int off = 16; off >= 1; off >>= 1) q += __shfl_xor(q, off, 32);
  float rs = rsqrtf(q * (1.0f / 256.0f) + 1e-5f);
#pragma unroll
  for (int i = 0; i < 8; ++i) {
    int c = lane * 8 + i;
    float y = (v[i] - mu) * rs * g[c] + b[c];
    if (outh) outh[row * ldoh + coloff + c] = (_Float16)y;
    if (outf) {
      long batch = row >> 12;
      long ip = row & 4095;
      outf[row * C + c] = y + xsrc[(batch * C + c) * N + ip];
    }
  }
}

// (b,c,4096) f32 -> (b,4096,c) f16 at row stride ldo
__global__ void transpose_f32_to_f16(const float* __restrict__ x,
                                     _Float16* __restrict__ out, int ldo) {
  long idx = (long)blockIdx.x * blockDim.x + threadIdx.x;
  long i = idx & 4095;
  long t = idx >> 12;
  long c = t & 255;
  long b = t >> 8;
  out[(b * 4096 + i) * ldo + c] = (_Float16)x[idx];
}

// W (KxN) f32 -> WT (NxK) f16
__global__ void convert_w_transpose(const float* __restrict__ w,
                                    _Float16* __restrict__ wt, int K, int N) {
  long idx = (long)blockIdx.x * blockDim.x + threadIdx.x;
  long k = idx / N;
  long j = idx - k * N;
  wt[j * K + k] = (_Float16)w[idx];
}

extern "C" void kernel_launch(void* const* d_in, const int* in_sizes, int n_in,
                              void* d_out, int out_size, void* d_ws, size_t ws_size,
                              hipStream_t stream) {
  (void)in_sizes; (void)n_in; (void)out_size; (void)ws_size;
  const float* x   = (const float*)d_in[0];
  const float* src = (const float*)d_in[1];
  const float* Wq  = (const float*)d_in[2];
  const float* bq  = (const float*)d_in[3];
  const float* Wk  = (const float*)d_in[4];
  const float* bk  = (const float*)d_in[5];
  const float* Wv  = (const float*)d_in[6];
  const float* bv  = (const float*)d_in[7];
  const float* Wm  = (const float*)d_in[8];
  const float* W1  = (const float*)d_in[9];
  const float* W2  = (const float*)d_in[10];
  const float* g1  = (const float*)d_in[11];
  const float* b1  = (const float*)d_in[12];
  const float* g2  = (const float*)d_in[13];
  const float* b2  = (const float*)d_in[14];
  float* out = (float*)d_out;

  char* ws = (char*)d_ws;
  size_t off = 0;
  auto alloc = [&](size_t bytes) -> void* {
    void* p = ws + off;
    off += (bytes + 255) & ~(size_t)255;
    return p;
  };
  const long B = 2, N = 4096, D = 256;
  _Float16* cat  = (_Float16*)alloc(B * N * 512 * 2);  // [xt_f16 | a_f16]
  _Float16* sth  = (_Float16*)alloc(B * N * D * 2);
  _Float16* qh   = (_Float16*)alloc(B * N * D * 2);
  _Float16* khb  = (_Float16*)alloc(B * N * D * 2);
  _Float16* vhb  = (_Float16*)alloc(B * N * D * 2);
  _Float16* atth = (_Float16*)alloc(B * N * D * 2);
  _Float16* h1   = (_Float16*)alloc(B * N * 512 * 2);
  float*    fbuf = (float*)   alloc(B * N * D * 4);
  _Float16* WqT  = (_Float16*)alloc(256 * 256 * 2);
  _Float16* WkT  = (_Float16*)alloc(256 * 256 * 2);
  _Float16* WvT  = (_Float16*)alloc(256 * 256 * 2);
  _Float16* WmT  = (_Float16*)alloc(256 * 256 * 2);
  _Float16* W1T  = (_Float16*)alloc(512 * 512 * 2);
  _Float16* W2T  = (_Float16*)alloc(512 * 256 * 2);

  dim3 blk(256);
  // transposed f16 activations
  transpose_f32_to_f16<<<dim3(8192), blk, 0, stream>>>(x, cat, 512);
  transpose_f32_to_f16<<<dim3(8192), blk, 0, stream>>>(src, sth, 256);
  // f16 transposed weights
  convert_w_transpose<<<dim3(256), blk, 0, stream>>>(Wq, WqT, 256, 256);
  convert_w_transpose<<<dim3(256), blk, 0, stream>>>(Wk, WkT, 256, 256);
  convert_w_transpose<<<dim3(256), blk, 0, stream>>>(Wv, WvT, 256, 256);
  convert_w_transpose<<<dim3(256), blk, 0, stream>>>(Wm, WmT, 256, 256);
  convert_w_transpose<<<dim3(1024), blk, 0, stream>>>(W1, W1T, 512, 512);
  convert_w_transpose<<<dim3(512), blk, 0, stream>>>(W2, W2T, 512, 256);

  dim3 g4(32, 4, 2);
  // Q = xt @ Wq + bq (xt is cols 0..255 of cat, lda=512)
  gemm_wmma_kernel<<<g4, blk, 0, stream>>>(cat, N * 512, 512, WqT, 256, bq,
                                           qh, N * 256, 256, nullptr, 0, 0, 256, 0);
  gemm_wmma_kernel<<<g4, blk, 0, stream>>>(sth, N * 256, 256, WkT, 256, bk,
                                           khb, N * 256, 256, nullptr, 0, 0, 256, 0);
  gemm_wmma_kernel<<<g4, blk, 0, stream>>>(sth, N * 256, 256, WvT, 256, bv,
                                           vhb, N * 256, 256, nullptr, 0, 0, 256, 0);

  flash_attn_kernel<<<dim3(32, 8, 2), blk, 0, stream>>>(qh, khb, vhb, atth);

  // merged = reshape(out) @ Wm  (reshape folded into attention store)
  gemm_wmma_kernel<<<g4, blk, 0, stream>>>(atth, N * 256, 256, WmT, 256, nullptr,
                                           nullptr, 0, 0, fbuf, N * 256, 256, 256, 0);
  // a = LN(merged) -> cat cols 256..511
  layernorm_kernel<<<dim3(1024), blk, 0, stream>>>(fbuf, g1, b1, cat, 512, 256,
                                                   nullptr, nullptr);
  // h1 = relu(cat @ Wmlp1)
  gemm_wmma_kernel<<<dim3(32, 8, 2), blk, 0, stream>>>(cat, N * 512, 512, W1T, 512,
                                                       nullptr, h1, N * 512, 512,
                                                       nullptr, 0, 0, 512, 1);
  // m = h1 @ Wmlp2
  gemm_wmma_kernel<<<g4, blk, 0, stream>>>(h1, N * 512, 512, W2T, 512, nullptr,
                                           nullptr, 0, 0, fbuf, N * 256, 256, 512, 0);
  // out = xt + LN(m)
  layernorm_kernel<<<dim3(1024), blk, 0, stream>>>(fbuf, g2, b2, nullptr, 0, 0,
                                                   out, x);
}
